// SelfAttentionSimple_27487790694978
// MI455X (gfx1250) — matrix operand order repaired and checked by reading, not compiled, not run on hardware
//
#include <hip/hip_runtime.h>
#include <math.h>

#define BB 4
#define SS 512
#define DD 256
#define BS (BB*SS)

typedef __attribute__((ext_vector_type(16))) __bf16 v16bf;
typedef __attribute__((ext_vector_type(8)))  __bf16 v8bf;
typedef __attribute__((ext_vector_type(8)))  float  v8f;

// Hardware tanh (V_TANH_F32 on gfx1250) with portable fallback.
#if defined(__has_builtin)
#if __has_builtin(__builtin_amdgcn_tanhf)
#define FAST_TANH(x) __builtin_amdgcn_tanhf(x)
#elif __has_builtin(__builtin_amdgcn_tanh_f32)
#define FAST_TANH(x) __builtin_amdgcn_tanh_f32(x)
#endif
#endif
#ifndef FAST_TANH
#define FAST_TANH(x) tanhf(x)
#endif

// ---- WMMA fragment helpers (per CDNA5 ISA 7.12.2 VGPR layouts, wave32) ----

// A matrix 16x32 bf16, row-major source with row stride lda (elements).
// lanes 0-15: M=lane, VGPR0-3 K=0..7, VGPR4-7 K=16..23
// lanes16-31: M=lane-16, VGPR0-3 K=8..15, VGPR4-7 K=24..31
__device__ __forceinline__ v16bf load_a_frag(const __bf16* A, int lda, int k0, int lane) {
    int row = lane & 15;
    int hi  = (lane >> 4) & 1;
    const __bf16* base = A + row * lda + k0 + hi * 8;
    v8bf lo = *(const v8bf*)(base);        // K = hi*8 .. hi*8+7
    v8bf hh = *(const v8bf*)(base + 16);   // K = 16+hi*8 .. 16+hi*8+7
    v16bf a;
#pragma unroll
    for (int i = 0; i < 8; ++i) { a[i] = lo[i]; a[i + 8] = hh[i]; }
    return a;
}

// B matrix 32x16 bf16 (K x N), supplied as transposed row-major BT[N][K]
// with row stride ldb, so each lane reads two runs of 8 contiguous K values.
// lanes 0-15: N=lane, K=0..7,16..23 ; lanes 16-31: N=lane-16, K=8..15,24..31
__device__ __forceinline__ v16bf load_bt_frag(const __bf16* BT, int ldb, int k0, int lane) {
    int col = lane & 15;
    int hi  = (lane >> 4) & 1;
    const __bf16* base = BT + col * ldb + k0 + hi * 8;
    v8bf lo = *(const v8bf*)(base);
    v8bf hh = *(const v8bf*)(base + 16);
    v16bf b;
#pragma unroll
    for (int i = 0; i < 8; ++i) { b[i] = lo[i]; b[i + 8] = hh[i]; }
    return b;
}

// C/D 16x16 f32: VGPR r -> row r (lanes 0-15) / row r+8 (lanes 16-31), col = lane&15
__device__ __forceinline__ void store_c_frag(float* C, int ldc, v8f acc, int lane) {
    int col   = lane & 15;
    int rbase = ((lane >> 4) & 1) * 8;
#pragma unroll
    for (int r = 0; r < 8; ++r) C[(rbase + r) * ldc + col] = acc[r];
}

// ---- Stage 0: dtype conversion / layout prep ----

__global__ __launch_bounds__(256) void convert_x_kernel(const float* __restrict__ x,
                                                        __bf16* __restrict__ xb,
                                                        __bf16* __restrict__ xT) {
    int e = blockIdx.x * 256 + threadIdx.x;       // e < BS*DD
    float v = x[e];
    __bf16 bv = (__bf16)v;
    xb[e] = bv;
    int r = e >> 8;            // row in [0, BS)
    int d = e & (DD - 1);
    int b = r >> 9;            // r / SS
    int j = r & (SS - 1);
    xT[((size_t)b * DD + d) * SS + j] = bv;       // xT[b][d][j]
}

__global__ __launch_bounds__(256) void convert_w_kernel(const float* __restrict__ wj,
                                                        const float* __restrict__ wt,
                                                        __bf16* __restrict__ wjT,
                                                        __bf16* __restrict__ wtT) {
    int e = blockIdx.x * 256 + threadIdx.x;       // e < DD*DD
    int k = e >> 8;
    int n = e & (DD - 1);
    wjT[n * DD + k] = (__bf16)wj[e];
    wtT[n * DD + k] = (__bf16)wt[e];
}

// ---- Stage 1: p1 = x@Wj, p2 = x@Wt  (shared A fragment, two WMMAs per K-step) ----

__global__ __launch_bounds__(256) void proj_gemm_kernel(const __bf16* __restrict__ xb,
                                                        const __bf16* __restrict__ wjT,
                                                        const __bf16* __restrict__ wtT,
                                                        float* __restrict__ p1,
                                                        float* __restrict__ p2) {
    int lane = threadIdx.x & 31;
    int w    = blockIdx.x * 8 + (threadIdx.x >> 5);   // global wave id, 2048 waves
    int m0   = (w >> 4) * 16;                          // row tile in [0, BS)
    int n0   = (w & 15) * 16;                          // col tile in [0, DD)
    const __bf16* A = xb + (size_t)m0 * DD;
    v8f accJ = {};
    v8f accT = {};
#pragma unroll
    for (int k0 = 0; k0 < DD; k0 += 32) {
        v16bf a  = load_a_frag(A, DD, k0, lane);
        v16bf bj = load_bt_frag(wjT + (size_t)n0 * DD, DD, k0, lane);
        v16bf bt = load_bt_frag(wtT + (size_t)n0 * DD, DD, k0, lane);
        accJ = __builtin_amdgcn_wmma_f32_16x16x32_bf16(false, a, false, bj, (short)0, accJ, false, false);
        accT = __builtin_amdgcn_wmma_f32_16x16x32_bf16(false, a, false, bt, (short)0, accT, false, false);
    }
    store_c_frag(p1 + (size_t)m0 * DD + n0, DD, accJ, lane);
    store_c_frag(p2 + (size_t)m0 * DD + n0, DD, accT, lane);
}

// ---- Stage 2: scores[b,i,j] = sum_d w[d]*tanh(p1[b,j,d] + p2[b,i,d]) ----
// One block per 16x16 (i,j) tile: 16x LDS reuse of each p1/p2 row.
// TRANS-bound: 1 V_TANH + 1 add + 1 FMA per element via hardware tanh;
// float4 LDS reads (rows stay 16B-aligned thanks to +4 pad) cut ds ops 4x.

__global__ __launch_bounds__(256) void scores_kernel(const float* __restrict__ p1,
                                                     const float* __restrict__ p2,
                                                     const float* __restrict__ wvec,
                                                     float* __restrict__ scores) {
    __shared__ float p1s[16][DD + 4];   // +4 pad: lane banks stagger by 4 -> conflict-free
    __shared__ float p2s[16][DD + 4];
    __shared__ float wsm[DD];
    int t  = threadIdx.x;
    int b  = blockIdx.z;
    int i0 = blockIdx.y * 16;
    int j0 = blockIdx.x * 16;
    const float* p1base = p1 + ((size_t)b * SS + j0) * DD;
    const float* p2base = p2 + ((size_t)b * SS + i0) * DD;
    for (int idx = t; idx < 16 * DD; idx += 256) {     // coalesced tile loads
        int r = idx >> 8, c = idx & (DD - 1);
        p1s[r][c] = p1base[r * DD + c];
        p2s[r][c] = p2base[r * DD + c];
    }
    if (t < DD) wsm[t] = wvec[t];
    __syncthreads();
    int j = t & 15, i = t >> 4;
    float acc = 0.0f;
#pragma unroll 4
    for (int d = 0; d < DD; d += 4) {
        float4 a4 = *(const float4*)&p1s[j][d];   // 16B ds_load, conflict-free
        float4 b4 = *(const float4*)&p2s[i][d];   // broadcast across j-lanes
        float4 w4 = *(const float4*)&wsm[d];
        acc += w4.x * FAST_TANH(a4.x + b4.x);
        acc += w4.y * FAST_TANH(a4.y + b4.y);
        acc += w4.z * FAST_TANH(a4.z + b4.z);
        acc += w4.w * FAST_TANH(a4.w + b4.w);
    }
    scores[((size_t)b * SS + i0 + i) * SS + j0 + j] = acc;
}

// ---- Stage 3: row softmax, emit bf16 probabilities ----

__global__ __launch_bounds__(256) void softmax_kernel(const float* __restrict__ scores,
                                                      __bf16* __restrict__ att) {
    __shared__ float red[256];
    int t = threadIdx.x;
    const float* row = scores + (size_t)blockIdx.x * SS;
    float v0 = row[t], v1 = row[t + 256];
    red[t] = fmaxf(v0, v1);
    __syncthreads();
    for (int s = 128; s > 0; s >>= 1) { if (t < s) red[t] = fmaxf(red[t], red[t + s]); __syncthreads(); }
    float M = red[0];
    __syncthreads();
    float e0 = __expf(v0 - M), e1 = __expf(v1 - M);
    red[t] = e0 + e1;
    __syncthreads();
    for (int s = 128; s > 0; s >>= 1) { if (t < s) red[t] += red[t + s]; __syncthreads(); }
    float inv = 1.0f / red[0];
    __bf16* orow = att + (size_t)blockIdx.x * SS;
    orow[t]       = (__bf16)(e0 * inv);
    orow[t + 256] = (__bf16)(e1 * inv);
}

// ---- Stage 4: c[b] = att[b] @ x[b]  (batched WMMA GEMM, f32 out) ----

__global__ __launch_bounds__(256) void ctx_gemm_kernel(const __bf16* __restrict__ att,
                                                       const __bf16* __restrict__ xT,
                                                       float* __restrict__ out) {
    int lane = threadIdx.x & 31;
    int w    = blockIdx.x * 8 + (threadIdx.x >> 5);   // 2048 waves
    int b    = w >> 9;                                 // 512 tiles per batch
    int tt   = w & 511;
    int m0   = (tt >> 4) * 16;                         // i tile in [0, SS)
    int n0   = (tt & 15) * 16;                         // d tile in [0, DD)
    const __bf16* A  = att + (size_t)b * SS * SS + (size_t)m0 * SS;  // [i][j], K=j contiguous
    const __bf16* BT = xT  + (size_t)b * DD * SS + (size_t)n0 * SS;  // [d][j], K=j contiguous
    v8f acc = {};
    for (int k0 = 0; k0 < SS; k0 += 32) {
        v16bf a  = load_a_frag(A, SS, k0, lane);
        v16bf bb = load_bt_frag(BT, SS, k0, lane);
        acc = __builtin_amdgcn_wmma_f32_16x16x32_bf16(false, a, false, bb, (short)0, acc, false, false);
    }
    store_c_frag(out + (size_t)b * SS * DD + (size_t)m0 * DD + n0, DD, acc, lane);
}

// ---- Host launch ----

extern "C" void kernel_launch(void* const* d_in, const int* in_sizes, int n_in,
                              void* d_out, int out_size, void* d_ws, size_t ws_size,
                              hipStream_t stream) {
    (void)in_sizes; (void)n_in; (void)out_size; (void)ws_size;
    const float* x  = (const float*)d_in[0];   // [B,S,D]
    const float* wv = (const float*)d_in[1];   // [1,D]
    const float* wj = (const float*)d_in[2];   // [D,D]
    const float* wt = (const float*)d_in[3];   // [D,D]
    float* out = (float*)d_out;                // [B,S,D]

    char* ws = (char*)d_ws;
    float*  p1     = (float*) (ws + 0);          // 2 MB
    float*  p2     = (float*) (ws + 2097152);    // 2 MB
    float*  scores = (float*) (ws + 4194304);    // 4 MB
    __bf16* xb     = (__bf16*)(ws + 8388608);    // 1 MB
    __bf16* xT     = (__bf16*)(ws + 9437184);    // 1 MB
    __bf16* wjT    = (__bf16*)(ws + 10485760);   // 128 KB
    __bf16* wtT    = (__bf16*)(ws + 10616832);   // 128 KB
    __bf16* att    = (__bf16*)(ws + 10747904);   // 2 MB

    convert_x_kernel<<<BS * DD / 256, 256, 0, stream>>>(x, xb, xT);
    convert_w_kernel<<<DD * DD / 256, 256, 0, stream>>>(wj, wt, wjT, wtT);
    proj_gemm_kernel<<<256, 256, 0, stream>>>(xb, wjT, wtT, p1, p2);
    dim3 sgrid(SS / 16, SS / 16, BB);
    scores_kernel<<<sgrid, 256, 0, stream>>>(p1, p2, wv, scores);
    softmax_kernel<<<BB * SS, 256, 0, stream>>>(scores, att);
    ctx_gemm_kernel<<<256, 256, 0, stream>>>(att, xT, out);
}
